// GCNBlock_17325898072380
// MI455X (gfx1250) — matrix operand order repaired and checked by reading, not compiled, not run on hardware
//
#include <hip/hip_runtime.h>
#include <hip/hip_bf16.h>

// Problem constants (from reference)
#define B_ 32
#define C_ 128
#define N_ 2025   // 45*45
#define K_ 9
#define G_ 4
#define CG_ (C_ / G_)   // 32 channels per group

typedef _Float16 v16h  __attribute__((ext_vector_type(16)));
typedef _Float16 f16x8 __attribute__((ext_vector_type(8)));
typedef float    v8f   __attribute__((ext_vector_type(8)));

// ---------------------------------------------------------------------------
// WMMA fragment loaders (CDNA5 16-bit layouts, CDNA5 ISA §7.12.2)
// A (16x32, MxK): lane holds row M=lane&15; VGPR0-3 -> K = kb + 8*half + {0..7}
//                 VGPR4-7 -> K = kb + 16 + 8*half + {0..7}
// B (32x16, KxN): lane holds col N=lane&15; same K mapping as A
// Both are contiguous runs in a row-major [*,128] f16 matrix -> b128 loads.
// ---------------------------------------------------------------------------
__device__ inline v16h load_a_frag(const _Float16* __restrict__ row, int kb, int hf) {
  f16x8 a0 = *(const f16x8*)(row + kb + hf * 8);
  f16x8 a1 = *(const f16x8*)(row + kb + 16 + hf * 8);
  v16h a;
#pragma unroll
  for (int i = 0; i < 8; ++i) { a[i] = a0[i]; a[i + 8] = a1[i]; }
  return a;
}

__device__ inline v16h load_b_frag(const _Float16* __restrict__ row, int kb, int hf) {
  f16x8 b0 = *(const f16x8*)(row + kb + hf * 8);
  f16x8 b1 = *(const f16x8*)(row + kb + 16 + hf * 8);
  v16h b;
#pragma unroll
  for (int i = 0; i < 8; ++i) { b[i] = b0[i]; b[i + 8] = b1[i]; }
  return b;
}

__device__ inline v8f wmma16(v16h a, v16h b, v8f c) {
  // 8 args: (neg_a, A, neg_b, B, c_mod, C, reuse_a, reuse_b)
  return __builtin_amdgcn_wmma_f32_16x16x32_f16(false, a, false, b, (short)0, c,
                                                false, false);
}

// ---------------------------------------------------------------------------
// 1) x[B,C,N] -> L2-normalized xn in f16, [B,N,C] row-major
// ---------------------------------------------------------------------------
__global__ __launch_bounds__(256) void k_normalize(const float* __restrict__ x,
                                                   _Float16* __restrict__ xnh) {
  int t = blockIdx.x * blockDim.x + threadIdx.x;
  if (t >= B_ * N_) return;
  int b = t / N_, n = t % N_;
  const float* xp = x + (size_t)b * C_ * N_ + n;
  float ss = 0.f;
#pragma unroll 8
  for (int c = 0; c < C_; ++c) { float v = xp[(size_t)c * N_]; ss += v * v; }
  float inv = 1.0f / fmaxf(sqrtf(ss), 1e-12f);
  _Float16* o = xnh + (size_t)t * C_;
#pragma unroll 8
  for (int c = 0; c < C_; ++c) o[c] = (_Float16)(xp[(size_t)c * N_] * inv);
}

// ---------------------------------------------------------------------------
// 2) Transpose + convert both weight matrices to f16: wt[c][k] = w[k][c]
// ---------------------------------------------------------------------------
__global__ __launch_bounds__(256) void k_wt(const float* __restrict__ w1,
                                            const float* __restrict__ w2,
                                            _Float16* __restrict__ w1t,
                                            _Float16* __restrict__ w2t) {
  int t = blockIdx.x * blockDim.x + threadIdx.x;
  if (t >= C_ * C_) return;
  int c = t / C_, k = t % C_;
  w1t[t] = (_Float16)w1[k * C_ + c];
  w2t[t] = (_Float16)w2[k * C_ + c];
}

// ---------------------------------------------------------------------------
// 3) Fused cosine-sim + top-9. One wave per (batch, 16-row tile).
//    Double-buffered B-fragment loads overlap the top-k merge; all 32 lanes
//    participate in the merge (8 columns each), reconciled at the end.
// ---------------------------------------------------------------------------
__global__ __launch_bounds__(32) void k_sim_topk(const _Float16* __restrict__ xnh,
                                                 float* __restrict__ vout,
                                                 int* __restrict__ iout) {
  __shared__ float tile[16][17];
  __shared__ float sv[16][K_];
  __shared__ int   si[16][K_];
  const int b = blockIdx.y;
  const int rowBase = blockIdx.x * 16;
  const int lane = threadIdx.x;
  const int hf = lane >> 4;
  const int l16 = lane & 15;
  const _Float16* xb = xnh + (size_t)b * N_ * C_;

  // A fragments: fixed for the whole block
  int mrow = rowBase + l16; if (mrow > N_ - 1) mrow = N_ - 1;
  const _Float16* arow = xb + (size_t)mrow * C_;
  v16h afr[4];
#pragma unroll
  for (int kk = 0; kk < 4; ++kk) afr[kk] = load_a_frag(arow, kk * 32, hf);

  // per-lane top-9 over its half-row (row = l16, 8 cols per tile)
  float tv[K_]; int ti[K_];
#pragma unroll
  for (int j = 0; j < K_; ++j) { tv[j] = -3.0e38f; ti[j] = 0; }
  float vmin = -3.0e38f; int pmin = 0;

  auto insert = [&](float val, int col) {
    if (val > vmin) {
      tv[pmin] = val; ti[pmin] = col;
      vmin = tv[0]; pmin = 0;
#pragma unroll
      for (int q = 1; q < K_; ++q)
        if (tv[q] < vmin) { vmin = tv[q]; pmin = q; }
    }
  };

  auto loadB = [&](int ct, v16h (&bfr)[4]) {
    int ncol = ct * 16 + l16; if (ncol > N_ - 1) ncol = N_ - 1;
    const _Float16* brow = xb + (size_t)ncol * C_;
#pragma unroll
    for (int kk = 0; kk < 4; ++kk) bfr[kk] = load_b_frag(brow, kk * 32, hf);
  };

  auto process = [&](int ct, v16h (&bfr)[4]) {
    v8f acc = {};
#pragma unroll
    for (int kk = 0; kk < 4; ++kk) acc = wmma16(afr[kk], bfr[kk], acc);
    // C/D layout: lanes 0-15 hold M=r, lanes 16-31 hold M=8+r, N=lane&15
#pragma unroll
    for (int r = 0; r < 8; ++r) tile[hf * 8 + r][l16] = acc[r];
    __syncthreads();
    // all 32 lanes merge: row = l16, cols = ct*16 + hf*8 + {0..7}
    const int colBase = ct * 16 + hf * 8;
#pragma unroll
    for (int j = 0; j < 8; ++j) {
      int col = colBase + j;
      if (col < N_) insert(tile[l16][hf * 8 + j], col);
    }
    __syncthreads();
  };

  const int numCT = (N_ + 15) / 16;  // 127
  v16h b0[4], b1[4];
  loadB(0, b0);
  for (int ct = 0; ct < numCT; ct += 2) {
    if (ct + 1 < numCT) loadB(ct + 1, b1);   // in flight during process(ct)
    process(ct, b0);
    if (ct + 1 < numCT) {
      if (ct + 2 < numCT) loadB(ct + 2, b0); // in flight during process(ct+1)
      process(ct + 1, b1);
    }
  }

  // reconcile the two half-row lists: high lanes publish, low lanes merge
  if (lane >= 16) {
#pragma unroll
    for (int j = 0; j < K_; ++j) { sv[l16][j] = tv[j]; si[l16][j] = ti[j]; }
  }
  __syncthreads();
  if (lane < 16) {
    int n = rowBase + lane;
    if (n < N_) {
#pragma unroll
      for (int j = 0; j < K_; ++j) insert(sv[l16][j], si[l16][j]);
      float s = 0.f;
#pragma unroll
      for (int j = 0; j < K_; ++j) s += tv[j];
      float inv = 1.0f / (s + 1e-6f);
      size_t o = ((size_t)b * N_ + n) * K_;
#pragma unroll
      for (int j = 0; j < K_; ++j) { vout[o + j] = tv[j] * inv; iout[o + j] = ti[j]; }
    }
  }
}

// ---------------------------------------------------------------------------
// 4) xt[b,n,:] = feat[b,n,:] @ W  (W pre-transposed f16), f32 output.
//    One wave per (batch, 16-row tile): 8 col tiles x 4 k-slabs = 32 WMMAs,
//    fragments batched ahead of the WMMA chain.
// ---------------------------------------------------------------------------
__global__ __launch_bounds__(32) void k_gemm_nc(const _Float16* __restrict__ feat,
                                                const _Float16* __restrict__ wt,
                                                float* __restrict__ out) {
  const int b = blockIdx.y;
  const int rowBase = blockIdx.x * 16;
  const int lane = threadIdx.x;
  const int hf = lane >> 4;
  const int l16 = lane & 15;
  const _Float16* fb = feat + (size_t)b * N_ * C_;

  int mrow = rowBase + l16; if (mrow > N_ - 1) mrow = N_ - 1;
  const _Float16* arow = fb + (size_t)mrow * C_;
  v16h afr[4];
#pragma unroll
  for (int kk = 0; kk < 4; ++kk) afr[kk] = load_a_frag(arow, kk * 32, hf);

#pragma unroll
  for (int ct = 0; ct < C_ / 16; ++ct) {
    const _Float16* brow = wt + (size_t)(ct * 16 + l16) * C_;
    v16h bfr[4];
#pragma unroll
    for (int kk = 0; kk < 4; ++kk) bfr[kk] = load_b_frag(brow, kk * 32, hf);
    v8f acc = {};
#pragma unroll
    for (int kk = 0; kk < 4; ++kk) acc = wmma16(afr[kk], bfr[kk], acc);
#pragma unroll
    for (int r = 0; r < 8; ++r) {
      int n = rowBase + hf * 8 + r;
      if (n < N_) out[((size_t)b * N_ + n) * C_ + ct * 16 + l16] = acc[r];
    }
  }
}

// ---------------------------------------------------------------------------
// 5) h[b,n,c] = bias[c] + sum_j v[b,n,j] * xt[b, idx[b,n,j], c]
// ---------------------------------------------------------------------------
__global__ __launch_bounds__(128) void k_gather(const float* __restrict__ xt,
                                                const float* __restrict__ v,
                                                const int* __restrict__ idx,
                                                const float* __restrict__ bias,
                                                float* __restrict__ h) {
  const int n = blockIdx.x, b = blockIdx.y, c = threadIdx.x;
  const size_t base = ((size_t)b * N_ + n) * K_;
  float acc = bias[c];
#pragma unroll
  for (int j = 0; j < K_; ++j) {
    float vj = v[base + j];
    int ij = idx[base + j];
    acc += vj * xt[((size_t)b * N_ + ij) * C_ + c];
  }
  h[((size_t)b * N_ + n) * C_ + c] = acc;
}

// ---------------------------------------------------------------------------
// 6) GroupNorm stats: one block per (group, batch). Deterministic tree reduce.
// ---------------------------------------------------------------------------
__global__ __launch_bounds__(256) void k_gn_stats(const float* __restrict__ h,
                                                  float* __restrict__ stats) {
  __shared__ float ssum[256], ssq[256];
  const int g = blockIdx.x, b = blockIdx.y, t = threadIdx.x;
  const int cnt = CG_ * N_;
  float s = 0.f, q = 0.f;
  for (int i = t; i < cnt; i += 256) {
    int n = i / CG_, cc = i % CG_;
    float v = h[((size_t)b * N_ + n) * C_ + g * CG_ + cc];
    s += v; q += v * v;
  }
  ssum[t] = s; ssq[t] = q;
  __syncthreads();
  for (int st = 128; st > 0; st >>= 1) {
    if (t < st) { ssum[t] += ssum[t + st]; ssq[t] += ssq[t + st]; }
    __syncthreads();
  }
  if (t == 0) {
    float mean = ssum[0] / (float)cnt;
    float var  = ssq[0] / (float)cnt - mean * mean;
    stats[(b * G_ + g) * 2 + 0] = mean;
    stats[(b * G_ + g) * 2 + 1] = rsqrtf(var + 1e-5f);
  }
}

// ---------------------------------------------------------------------------
// 7a) GN + SiLU -> f16 features for next layer (same [B,N,C] layout)
// ---------------------------------------------------------------------------
__global__ __launch_bounds__(256) void k_gn_silu_mid(const float* __restrict__ h,
                                                     const float* __restrict__ stats,
                                                     const float* __restrict__ gamma,
                                                     const float* __restrict__ beta,
                                                     _Float16* __restrict__ ofh) {
  int t = blockIdx.x * blockDim.x + threadIdx.x;
  if (t >= B_ * N_ * C_) return;
  int c = t % C_;
  int b = t / (N_ * C_);
  int g = c / CG_;
  float m  = stats[(b * G_ + g) * 2 + 0];
  float rs = stats[(b * G_ + g) * 2 + 1];
  float y = (h[t] - m) * rs * gamma[c] + beta[c];
  float sy = y / (1.0f + __expf(-y));
  ofh[t] = (_Float16)sy;
}

// ---------------------------------------------------------------------------
// 7b) Final GN + SiLU, transposing [B,N,C] -> [B,C,H,W] f32 output
// ---------------------------------------------------------------------------
__global__ __launch_bounds__(256) void k_gn_silu_out(const float* __restrict__ h,
                                                     const float* __restrict__ stats,
                                                     const float* __restrict__ gamma,
                                                     const float* __restrict__ beta,
                                                     float* __restrict__ out) {
  int t = blockIdx.x * blockDim.x + threadIdx.x;
  if (t >= B_ * N_ * C_) return;
  int c = t % C_;
  int bn = t / C_;
  int n = bn % N_;
  int b = bn / N_;
  int g = c / CG_;
  float m  = stats[(b * G_ + g) * 2 + 0];
  float rs = stats[(b * G_ + g) * 2 + 1];
  float y = (h[t] - m) * rs * gamma[c] + beta[c];
  float sy = y / (1.0f + __expf(-y));
  out[((size_t)b * C_ + c) * N_ + n] = sy;
}

// ---------------------------------------------------------------------------
extern "C" void kernel_launch(void* const* d_in, const int* in_sizes, int n_in,
                              void* d_out, int out_size, void* d_ws, size_t ws_size,
                              hipStream_t stream) {
  (void)in_sizes; (void)n_in; (void)out_size; (void)ws_size;
  const float* x      = (const float*)d_in[0];
  const float* w1     = (const float*)d_in[1];
  const float* b1     = (const float*)d_in[2];
  const float* w2     = (const float*)d_in[3];
  const float* b2     = (const float*)d_in[4];
  const float* gamma1 = (const float*)d_in[5];
  const float* beta1  = (const float*)d_in[6];
  const float* gamma2 = (const float*)d_in[7];
  const float* beta2  = (const float*)d_in[8];
  float* out = (float*)d_out;

  char* p = (char*)d_ws;
  auto carve = [&](size_t bytes) -> char* {
    char* r = p;
    p += (bytes + 255) & ~(size_t)255;
    return r;
  };
  _Float16* xnh   = (_Float16*)carve((size_t)B_ * N_ * C_ * sizeof(_Float16));
  _Float16* ofh   = (_Float16*)carve((size_t)B_ * N_ * C_ * sizeof(_Float16));
  _Float16* w1t   = (_Float16*)carve((size_t)C_ * C_ * sizeof(_Float16));
  _Float16* w2t   = (_Float16*)carve((size_t)C_ * C_ * sizeof(_Float16));
  float*    vbuf  = (float*)carve((size_t)B_ * N_ * K_ * sizeof(float));
  int*      ibuf  = (int*)carve((size_t)B_ * N_ * K_ * sizeof(int));
  float*    xt    = (float*)carve((size_t)B_ * N_ * C_ * sizeof(float));
  float*    hbuf  = (float*)carve((size_t)B_ * N_ * C_ * sizeof(float));
  float*    stats = (float*)carve((size_t)B_ * G_ * 2 * sizeof(float));

  const int rowTiles = (N_ + 15) / 16;
  const int nElem = B_ * N_ * C_;

  k_wt<<<(C_ * C_ + 255) / 256, 256, 0, stream>>>(w1, w2, w1t, w2t);
  k_normalize<<<(B_ * N_ + 255) / 256, 256, 0, stream>>>(x, xnh);
  k_sim_topk<<<dim3(rowTiles, B_), 32, 0, stream>>>(xnh, vbuf, ibuf);

  // Layer 1
  k_gemm_nc<<<dim3(rowTiles, B_), 32, 0, stream>>>(xnh, w1t, xt);
  k_gather<<<dim3(N_, B_), 128, 0, stream>>>(xt, vbuf, ibuf, b1, hbuf);
  k_gn_stats<<<dim3(G_, B_), 256, 0, stream>>>(hbuf, stats);
  k_gn_silu_mid<<<(nElem + 255) / 256, 256, 0, stream>>>(hbuf, stats, gamma1, beta1, ofh);

  // Layer 2
  k_gemm_nc<<<dim3(rowTiles, B_), 32, 0, stream>>>(ofh, w2t, xt);
  k_gather<<<dim3(N_, B_), 128, 0, stream>>>(xt, vbuf, ibuf, b2, hbuf);
  k_gn_stats<<<dim3(G_, B_), 256, 0, stream>>>(hbuf, stats);
  k_gn_silu_out<<<(nElem + 255) / 256, 256, 0, stream>>>(hbuf, stats, gamma2, beta2, out);
}
